// DGCNNEncoder_26963804685129
// MI455X (gfx1250) — matrix operand order repaired and checked by reading, not compiled, not run on hardware
//
#include <hip/hip_runtime.h>

#define B_   8
#define N_   2048
#define KNN  20

typedef __bf16 bf16_t;
typedef bf16_t v16bf __attribute__((ext_vector_type(16)));
typedef float  v8f   __attribute__((ext_vector_type(8)));

// ---------- helpers ----------

__device__ __forceinline__ bf16_t f2bf(float f) {
  unsigned u = __builtin_bit_cast(unsigned, f);
  unsigned r = (u + 0x7FFFu + ((u >> 16) & 1u)) >> 16;
  unsigned short s = (unsigned short)r;
  return __builtin_bit_cast(bf16_t, s);
}

__device__ __forceinline__ v8f wmma_bf16(v16bf a, v16bf b, v8f c) {
  // D = A(16x32) * B(32x16) + C, f32 accumulate
  return __builtin_amdgcn_wmma_f32_16x16x32_bf16(false, a, false, b, (short)0, c,
                                                 false, false);
}

// A fragment (16x32): bf16 LDS source, row-major, stride LD (elements).
// lane m, half h reads K = 16h..16h+15 contiguous (matches B fragments).
// Two b128 LDS loads, no VALU.
template <int LD>
__device__ __forceinline__ v16bf load_a_bf(const bf16_t* base, int k0, int lane) {
  int m = lane & 15, half = lane >> 4;
  const bf16_t* p = base + m * LD + k0;
  v16bf a;
#pragma unroll
  for (int e = 0; e < 8; ++e) a[e] = p[16 * half + e];
#pragma unroll
  for (int e = 0; e < 8; ++e) a[8 + e] = p[16 * half + 8 + e];
  return a;
}

// B fragment (32x16): element (k, n) = base[n*LD + k]; lane n, half h reads
// K = 16h..16h+15 contiguous.  Two b128 LDS loads, no VALU.
template <int LD>
__device__ __forceinline__ v16bf load_b_bf(const bf16_t* base, int k0, int lane) {
  int n = lane & 15, half = lane >> 4;
  const bf16_t* p = base + n * LD + k0 + 16 * half;
  v16bf b;
#pragma unroll
  for (int e = 0; e < 16; ++e) b[e] = p[e];
  return b;
}

// ---------- kNN: WMMA distance tiles + per-row top-20 ----------

template <int C>
__global__ __launch_bounds__(128) void knn_kernel(const float* __restrict__ X,
                                                  int* __restrict__ idx) {
  constexpr int CP = (C + 31) & ~31;  // K padded to 32
  constexpr int NKC = CP / 32;
  constexpr int QT = 128, CT = 128;
  __shared__ bf16_t qf[QT * CP];
  __shared__ bf16_t cf[CT * CP];
  __shared__ float qn[QT];
  __shared__ float cn[CT];
  __shared__ float d2s[QT * CT];

  const int b = blockIdx.y;
  const int q0 = blockIdx.x * QT;
  const int tid = threadIdx.x;
  const int lane = tid & 31, wv = tid >> 5;

  { // one query row per thread; convert to bf16 once
    const float* src = X + ((long)b * N_ + q0 + tid) * C;
    float s = 0.f;
    for (int c = 0; c < CP; ++c) {
      float v = (c < C) ? src[c] : 0.f;
      qf[tid * CP + c] = f2bf(v);
      s += v * v;
    }
    qn[tid] = s;
  }

  float bd[KNN];
  int bi[KNN];
#pragma unroll
  for (int k = 0; k < KNN; ++k) { bd[k] = 3.0e38f; bi[k] = 0; }

  for (int c0 = 0; c0 < N_; c0 += CT) {
    __syncthreads(); // previous d2s consumed / cf free
    {
      const float* src = X + ((long)b * N_ + c0 + tid) * C;
      float s = 0.f;
      for (int c = 0; c < CP; ++c) {
        float v = (c < C) ? src[c] : 0.f;
        cf[tid * CP + c] = f2bf(v);
        s += v * v;
      }
      cn[tid] = s;
    }
    __syncthreads();

    // wave owns col-tiles {wv, wv+4}; B fragments register-resident per chunk
    v16bf bfr[2][NKC];
#pragma unroll
    for (int h = 0; h < 2; ++h)
#pragma unroll
      for (int kc = 0; kc < NKC; ++kc)
        bfr[h][kc] = load_b_bf<CP>(cf + (wv + 4 * h) * 16 * CP, kc * 32, lane);

    const int col0 = wv * 16 + (lane & 15);
    const int col1 = (wv + 4) * 16 + (lane & 15);
#pragma unroll 2
    for (int ti = 0; ti < 8; ++ti) {
      v8f acc0 = {}, acc1 = {};
#pragma unroll
      for (int kc = 0; kc < NKC; ++kc) {
        v16bf a = load_a_bf<CP>(qf + ti * 16 * CP, kc * 32, lane);
        acc0 = wmma_bf16(a, bfr[0][kc], acc0);
        acc1 = wmma_bf16(a, bfr[1][kc], acc1);
      }
      int rbase = ti * 16 + ((lane < 16) ? 0 : 8);
#pragma unroll
      for (int r = 0; r < 8; ++r) {
        int row = rbase + r;
        d2s[row * CT + col0] = qn[row] + cn[col0] - 2.0f * acc0[r];
        d2s[row * CT + col1] = qn[row] + cn[col1] - 2.0f * acc1[r];
      }
    }
    __syncthreads();

    // top-20 insertion, thread owns query row `tid`
    for (int j = 0; j < CT; ++j) {
      float d = d2s[tid * CT + j];
      if (d < bd[KNN - 1]) {
        int p = KNN - 1;
        while (p > 0 && bd[p - 1] > d) {
          bd[p] = bd[p - 1];
          bi[p] = bi[p - 1];
          --p;
        }
        bd[p] = d;
        bi[p] = c0 + j;
      }
    }
  }

  int* dst = idx + ((long)b * N_ + q0 + tid) * KNN;
#pragma unroll
  for (int k = 0; k < KNN; ++k) dst[k] = bi[k];
}

// ---------- EdgeConv: LDS edge tensor (bf16) -> WMMA -> leaky -> max over k ----

template <int CIN, int COUT>
__global__ __launch_bounds__(128) void edgeconv_kernel(
    const float* __restrict__ Xin, const float* __restrict__ W,
    const float* __restrict__ g, const float* __restrict__ bias,
    const int* __restrict__ idx, float* __restrict__ Xout) {
  constexpr int PT = 16;               // points per block
  constexpr int K2 = 2 * CIN;          // GEMM K dim (pre-pad)
  constexpr int CP2 = (K2 + 31) & ~31; // padded to 32
  constexpr int NK = CP2 / 32;
  __shared__ float ctr[PT * CIN];          // centers, f32 (for subtraction)
  __shared__ bf16_t edg[KNN * PT * CP2];   // edge features, bf16, zero-padded

  const int b = blockIdx.y;
  const int n0 = blockIdx.x * PT;
  const int tid = threadIdx.x;
  const int lane = tid & 31, wv = tid >> 5;

  if (tid < PT) {
    const float* src = Xin + ((long)b * N_ + n0 + tid) * CIN;
    for (int c = 0; c < CIN; ++c) ctr[tid * CIN + c] = src[c];
  }
  __syncthreads();

  // build edge tensor once: [nbr - ctr | ctr | 0-pad], bf16
  for (int p = tid; p < KNN * PT; p += 128) {
    int k = p / PT, m = p % PT;
    int nb = idx[((long)b * N_ + n0 + m) * KNN + k];
    const float* src = Xin + ((long)b * N_ + nb) * CIN;
    bf16_t* dst = edg + (k * PT + m) * CP2;
    for (int c = 0; c < CIN; ++c) dst[c] = f2bf(src[c] - ctr[m * CIN + c]);
    for (int c = 0; c < CIN; ++c) dst[CIN + c] = f2bf(ctr[m * CIN + c]);
    for (int c = K2; c < CP2; ++c) dst[c] = f2bf(0.f);
  }
  __syncthreads();

  for (int ct = wv; ct < COUT / 16; ct += 4) {
    const int o0 = ct * 16;

    // preload W col-tile as B fragments (converted once, zero-padded past K2)
    v16bf wf[NK];
#pragma unroll
    for (int kc = 0; kc < NK; ++kc) {
      int n = lane & 15, half = lane >> 4;
      v16bf bb;
#pragma unroll
      for (int e = 0; e < 16; ++e) {
        int kk = kc * 32 + 16 * half + e;
        bb[e] = f2bf(kk < K2 ? W[(long)(o0 + n) * K2 + kk] : 0.f);
      }
      wf[kc] = bb;
    }

    const float gg = g[o0 + (lane & 15)];
    const float bt = bias[o0 + (lane & 15)];
    v8f mx;
#pragma unroll
    for (int r = 0; r < 8; ++r) mx[r] = -3.0e38f;

    for (int k = 0; k < KNN; ++k) {
      v8f acc = {};
#pragma unroll
      for (int kc = 0; kc < NK; ++kc) {
        v16bf a = load_a_bf<CP2>(edg + k * PT * CP2, kc * 32, lane);
        acc = wmma_bf16(a, wf[kc], acc);
      }
#pragma unroll
      for (int r = 0; r < 8; ++r) {
        float h = acc[r] * gg + bt;
        h = (h >= 0.f) ? h : 0.2f * h;
        mx[r] = fmaxf(mx[r], h);
      }
    }

    const int o = o0 + (lane & 15);
    const int rbase = (lane < 16) ? 0 : 8;
#pragma unroll
    for (int r = 0; r < 8; ++r) {
      int m = rbase + r;
      Xout[((long)b * N_ + n0 + m) * COUT + o] = mx[r];
    }
  }
}

// ---------- final projection + fused max/mean over N ----------

__global__ __launch_bounds__(128) void final_kernel(
    const float* __restrict__ x1, const float* __restrict__ x2,
    const float* __restrict__ x3, const float* __restrict__ x4,
    const float* __restrict__ W5, const float* __restrict__ g5,
    const float* __restrict__ b5, float* __restrict__ out) {
  constexpr int CK = 512;
  __shared__ bf16_t xc[16 * CK]; // 16 KB x_cat tile (bf16)

  const int b = blockIdx.y;
  const int tid = threadIdx.x;
  const int lane = tid & 31, wv = tid >> 5;
  const int o0 = (blockIdx.x * 4 + wv) * 16;

  // preload W5 col-tile: 16 B fragments, register-resident for the whole kernel
  v16bf wf[16];
#pragma unroll
  for (int kc = 0; kc < 16; ++kc) {
    int n = lane & 15, half = lane >> 4;
    const float* p = W5 + (long)(o0 + n) * CK + kc * 32 + 16 * half;
    v16bf bb;
#pragma unroll
    for (int e = 0; e < 16; ++e) bb[e] = f2bf(p[e]);
    wf[kc] = bb;
  }

  const float gg = g5[o0 + (lane & 15)];
  const float bt = b5[o0 + (lane & 15)];
  float runmax = -3.0e38f, runsum = 0.f;

  for (int n0 = 0; n0 < N_; n0 += 16) {
    __syncthreads();
    for (int p = tid; p < 16 * CK; p += 128) {
      int m = p / CK, c = p % CK;
      long r = (long)b * N_ + n0 + m;
      float v;
      if (c < 64)       v = x1[r * 64 + c];
      else if (c < 128) v = x2[r * 64 + (c - 64)];
      else if (c < 256) v = x3[r * 128 + (c - 128)];
      else              v = x4[r * 256 + (c - 256)];
      xc[m * CK + c] = f2bf(v);
    }
    __syncthreads();

    v8f acc = {};
#pragma unroll
    for (int kc = 0; kc < 16; ++kc) {
      v16bf a = load_a_bf<CK>(xc, kc * 32, lane);
      acc = wmma_bf16(a, wf[kc], acc);
    }
#pragma unroll
    for (int r = 0; r < 8; ++r) {
      float h = acc[r] * gg + bt;
      h = (h >= 0.f) ? h : 0.2f * h;
      runmax = fmaxf(runmax, h);
      runsum += h;
    }
  }

  float om = fmaxf(runmax, __shfl_xor(runmax, 16, 32));
  float os = runsum + __shfl_xor(runsum, 16, 32);
  if (lane < 16) {
    out[(long)b * 2048 + o0 + lane] = om;
    out[(long)b * 2048 + 1024 + o0 + lane] = os * (1.0f / N_);
  }
}

// ---------- launch ----------

extern "C" void kernel_launch(void* const* d_in, const int* in_sizes, int n_in,
                              void* d_out, int out_size, void* d_ws, size_t ws_size,
                              hipStream_t stream) {
  (void)in_sizes; (void)n_in; (void)out_size; (void)ws_size;
  const float* x  = (const float*)d_in[0];
  const float* W1 = (const float*)d_in[1];
  const float* W2 = (const float*)d_in[2];
  const float* W3 = (const float*)d_in[3];
  const float* W4 = (const float*)d_in[4];
  const float* W5 = (const float*)d_in[5];
  const float* g1 = (const float*)d_in[6];
  const float* b1 = (const float*)d_in[7];
  const float* g2 = (const float*)d_in[8];
  const float* b2 = (const float*)d_in[9];
  const float* g3 = (const float*)d_in[10];
  const float* b3 = (const float*)d_in[11];
  const float* g4 = (const float*)d_in[12];
  const float* b4 = (const float*)d_in[13];
  const float* g5 = (const float*)d_in[14];
  const float* b5 = (const float*)d_in[15];
  float* out = (float*)d_out;

  // workspace layout: x1|x2|x3|x4|idx  (~34 MB, fits easily in L2's 192 MB)
  float* x1 = (float*)d_ws;
  float* x2 = x1 + (size_t)B_ * N_ * 64;
  float* x3 = x2 + (size_t)B_ * N_ * 64;
  float* x4 = x3 + (size_t)B_ * N_ * 128;
  int* idxb = (int*)(x4 + (size_t)B_ * N_ * 256);

  dim3 blk(128);
  dim3 gknn(N_ / 128, B_);
  dim3 gec(N_ / 16, B_);
  dim3 gfin(1024 / 64, B_);

  knn_kernel<3><<<gknn, blk, 0, stream>>>(x, idxb);
  edgeconv_kernel<3, 64><<<gec, blk, 0, stream>>>(x, W1, g1, b1, idxb, x1);

  knn_kernel<64><<<gknn, blk, 0, stream>>>(x1, idxb);
  edgeconv_kernel<64, 64><<<gec, blk, 0, stream>>>(x1, W2, g2, b2, idxb, x2);

  knn_kernel<64><<<gknn, blk, 0, stream>>>(x2, idxb);
  edgeconv_kernel<64, 128><<<gec, blk, 0, stream>>>(x2, W3, g3, b3, idxb, x3);

  knn_kernel<128><<<gknn, blk, 0, stream>>>(x3, idxb);
  edgeconv_kernel<128, 256><<<gec, blk, 0, stream>>>(x3, W4, g4, b4, idxb, x4);

  final_kernel<<<gfin, blk, 0, stream>>>(x1, x2, x3, x4, W5, g5, b5, out);
}
